// MLPEncoder_17025250361877
// MI455X (gfx1250) — compile-verified, matrix operand order checked by reading, not run on hardware
//
#include <hip/hip_runtime.h>

// ---------------------------------------------------------------------------
// GATv2 x2 encoder for gfx1250 (MI455X).
// Pipeline:
//   1. convert 4x [256x256] fp32 weights -> bf16, transposed [n][k]
//   2. GEMM (bf16 WMMA 16x16x32, f32 accum): xl = x@W1l, xr = x@W1r
//   3. edge phase: score(+atomicMax) -> exp(+atomicAdd den) -> aggregate
//   4. BatchNorm(+affine)+ReLU (in-place on agg)
//   5. GEMM: xl = h@W2l, xr = h@W2r ; edge phase again (same dims)
//   6. head-mean (+b2), BatchNorm(no affine) on cols 0..31, sigmoid on 32..63
// ---------------------------------------------------------------------------

typedef __attribute__((ext_vector_type(16))) __bf16 v16bf;
typedef __attribute__((ext_vector_type(8)))  __bf16 v8bf;
typedef __attribute__((ext_vector_type(8)))  float  v8f;
typedef __attribute__((ext_vector_type(4)))  float  v4f;

#define F   256   // feature dim everywhere (IN = H*HID = H*2*OUT = 256)
#define HH  4     // heads
#define CH  64    // channels per head in both convs

// ---------------- helpers ----------------

static __device__ __forceinline__ unsigned f2ord(float f) {
  unsigned u = __float_as_uint(f);
  return (u & 0x80000000u) ? ~u : (u | 0x80000000u);   // monotone float->uint
}
static __device__ __forceinline__ float ord2f(unsigned o) {
  unsigned u = (o & 0x80000000u) ? (o & 0x7fffffffu) : ~o;
  return __uint_as_float(u);
}
static __device__ __forceinline__ void atomAddF(float* p, float v) {
  __hip_atomic_fetch_add(p, v, __ATOMIC_RELAXED, __HIP_MEMORY_SCOPE_AGENT);
}
static __device__ __forceinline__ void atomMaxU(unsigned* p, unsigned v) {
  __hip_atomic_fetch_max(p, v, __ATOMIC_RELAXED, __HIP_MEMORY_SCOPE_AGENT);
}

// A fragment (16x32, M x K, bf16 from fp32 source), ISA 16-bit A layout:
// lanes 0-15: M=lane,   K = k0 + {0..7, 16..23}
// lanes16-31: M=lane-16,K = k0 + {8..15, 24..31}
// caller pre-offsets p to row (m0 + lane&15), col k0 + ((lane>>4)<<3)
static __device__ __forceinline__ v16bf a_frag(const float* __restrict__ p) {
  const v4f* q = (const v4f*)p;
  v4f a0 = q[0], a1 = q[1], a2 = q[4], a3 = q[5];
  v16bf r;
#pragma unroll
  for (int j = 0; j < 4; ++j) {
    r[j]      = (__bf16)a0[j];
    r[4 + j]  = (__bf16)a1[j];
    r[8 + j]  = (__bf16)a2[j];
    r[12 + j] = (__bf16)a3[j];
  }
  return r;
}

// B fragment (32x16, K x N): B stored transposed [n][k] bf16; by symmetry the
// lane/element -> (N,K) map is identical to A's (N = lane&15). Two 16B loads.
static __device__ __forceinline__ v16bf b_frag(const __bf16* __restrict__ p) {
  v8bf lo = *(const v8bf*)p;
  v8bf hi = *(const v8bf*)(p + 16);
  return __builtin_shufflevector(lo, hi, 0, 1, 2, 3, 4, 5, 6, 7,
                                 8, 9, 10, 11, 12, 13, 14, 15);
}

// ---------------- kernels ----------------

// W [k][n] fp32 -> BT [n][k] bf16   (256x256)
__global__ __launch_bounds__(256) void wconvert(const float* __restrict__ W,
                                                __bf16* __restrict__ BT) {
  int idx = blockIdx.x * 256 + threadIdx.x;     // 65536 threads
  int n = idx >> 8, k = idx & 255;
  BT[idx] = (__bf16)W[(size_t)k * F + n];
}

// C[M x 256] = A[M x 256] * B, B given as BT[n][k] bf16.
// One wave per 16-row strip: A K-fragments held in VGPRs, sweep 16 N-tiles,
// 8 x v_wmma_f32_16x16x32_bf16 each.
__global__ __launch_bounds__(256) void gemm_bf16(const float* __restrict__ A,
                                                 const __bf16* __restrict__ BT,
                                                 float* __restrict__ C, int M) {
  const int lane = threadIdx.x & 31;
  const int m0 = (blockIdx.x * 8 + (threadIdx.x >> 5)) * 16;
  if (m0 >= M) return;                       // uniform per wave: EXEC stays all-1
  const int mr = lane & 15;
  const int kb = (lane >> 4) << 3;
  const float* arow = A + (size_t)(m0 + mr) * F + kb;
  v16bf afr[8];
#pragma unroll
  for (int kk = 0; kk < 8; ++kk) afr[kk] = a_frag(arow + kk * 32);
  const __bf16* bbase = BT + (size_t)mr * F + kb;
  for (int nt = 0; nt < 16; ++nt) {
    const __bf16* brow = bbase + (size_t)(nt * 16) * F;
    v8f acc = {};
#pragma unroll
    for (int kk = 0; kk < 8; ++kk) {
      v16bf bf = b_frag(brow + kk * 32);
      acc = __builtin_amdgcn_wmma_f32_16x16x32_bf16(false, afr[kk], false, bf,
                                                    (short)0, acc, false, false);
    }
    // D layout: lane -> N = lane&15, VGPR i -> M = m0 + kb + i
    const int n = nt * 16 + mr;
    float* crow = C + (size_t)(m0 + kb) * F + n;
#pragma unroll
    for (int i = 0; i < 8; ++i) crow[(size_t)i * F] = acc[i];
  }
}

// One wave per edge: score[e][h] = att[h] . leaky_relu(xl[src]+xr[dst]) and
// running segment max per (dst, head).
__global__ __launch_bounds__(256) void gat_score(
    const float* __restrict__ xl, const float* __restrict__ xr,
    const float* __restrict__ att, const int* __restrict__ ei,
    int E, int Etot, float* __restrict__ score, unsigned* __restrict__ nmax) {
  const int e = blockIdx.x * 8 + (threadIdx.x >> 5);
  if (e >= Etot) return;
  const int lane = threadIdx.x & 31;
  int src, dst;
  if (e < E) { src = ei[e]; dst = ei[E + e]; } else { src = dst = e - E; }
  const float* ps = xl + (size_t)src * F;
  const float* pd = xr + (size_t)dst * F;
  float sc[HH];
#pragma unroll
  for (int h = 0; h < HH; ++h) {
    float p = 0.f;
#pragma unroll
    for (int s = 0; s < 2; ++s) {
      int c = h * CH + s * 32 + lane;           // coalesced 128B per step
      float v = ps[c] + pd[c];
      v = (v > 0.f) ? v : v * 0.2f;             // leaky_relu 0.2
      p += v * att[c];
    }
#pragma unroll
    for (int off = 16; off > 0; off >>= 1) p += __shfl_xor(p, off, 32);
    sc[h] = p;
  }
  if (lane == 0) {
#pragma unroll
    for (int h = 0; h < HH; ++h) {
      score[(size_t)e * HH + h] = sc[h];
      atomMaxU(&nmax[dst * HH + h], f2ord(sc[h]));
    }
  }
}

// One thread per (edge, head): ex = exp(score - max[dst]); den[dst] += ex.
__global__ __launch_bounds__(256) void gat_expsum(
    const int* __restrict__ ei, int E, int Etot,
    const unsigned* __restrict__ nmax, float* __restrict__ score,
    float* __restrict__ den) {
  int idx = blockIdx.x * 256 + threadIdx.x;
  if (idx >= Etot * HH) return;
  int e = idx >> 2, h = idx & 3;
  int dst = (e < E) ? ei[E + e] : (e - E);
  float m = ord2f(nmax[dst * HH + h]);
  float ex = __expf(score[idx] - m);
  score[idx] = ex;                               // overwrite with exp
  atomAddF(&den[dst * HH + h], ex);
}

// One wave per edge: out[dst] += alpha * xl[src] over 256 channels.
__global__ __launch_bounds__(256) void gat_aggregate(
    const int* __restrict__ ei, int E, int Etot,
    const float* __restrict__ xl, const float* __restrict__ score,
    const float* __restrict__ den, float* __restrict__ out) {
  const int e = blockIdx.x * 8 + (threadIdx.x >> 5);
  if (e >= Etot) return;
  const int lane = threadIdx.x & 31;
  int src, dst;
  if (e < E) { src = ei[e]; dst = ei[E + e]; } else { src = dst = e - E; }
  float al[HH];
#pragma unroll
  for (int h = 0; h < HH; ++h)
    al[h] = score[(size_t)e * HH + h] / den[dst * HH + h];
  const float* ps = xl + (size_t)src * F;
  float* po = out + (size_t)dst * F;
#pragma unroll
  for (int s = 0; s < 8; ++s) {
    int c = s * 32 + lane;
    atomAddF(po + c, al[s >> 1] * ps[c]);        // head = c>>6 = s>>1
  }
}

// Per-column partial sums (coalesced: thread t owns column t of its block's
// row strip). blockDim.x == C.
__global__ void bn_partial(const float* __restrict__ X, int rows, int C,
                           float* __restrict__ sum, float* __restrict__ ssq) {
  int col = threadIdx.x;
  float s = 0.f, s2 = 0.f;
  for (int r = blockIdx.x; r < rows; r += gridDim.x) {
    float v = X[(size_t)r * C + col];
    s += v; s2 += v * v;
  }
  atomAddF(&sum[col], s);
  atomAddF(&ssq[col], s2);
}

__global__ void bn_finalize(const float* __restrict__ sum,
                            const float* __restrict__ ssq, int rows, int C,
                            float* __restrict__ mu, float* __restrict__ rs) {
  int c = threadIdx.x;
  if (c >= C) return;
  float m = sum[c] / (float)rows;
  float var = ssq[c] / (float)rows - m * m;      // biased, as in BN training
  mu[c] = m;
  rs[c] = rsqrtf(var + 1e-5f);
}

// h = relu(gamma*(x-mu)*rsig + beta), in place. (conv1 bias b1 cancels in BN.)
__global__ __launch_bounds__(256) void bn_apply_relu(
    float* __restrict__ X, int total, const float* __restrict__ mu,
    const float* __restrict__ rs, const float* __restrict__ g,
    const float* __restrict__ be) {
  int idx = blockIdx.x * 256 + threadIdx.x;
  if (idx >= total) return;
  int c = idx & 255;
  float v = (X[idx] - mu[c]) * rs[c] * g[c] + be[c];
  X[idx] = v > 0.f ? v : 0.f;
}

// o[n][c] = mean over heads of agg[n][h*64+c] + b2[c]
__global__ __launch_bounds__(256) void head_mean(
    const float* __restrict__ agg, const float* __restrict__ b2,
    float* __restrict__ o, int total) {
  int idx = blockIdx.x * 256 + threadIdx.x;
  if (idx >= total) return;
  int r = idx >> 6, c = idx & 63;
  const float* p = agg + (size_t)r * F;
  o[idx] = 0.25f * (p[c] + p[CH + c] + p[2 * CH + c] + p[3 * CH + c]) + b2[c];
}

// z_mu = (o[:, :32]-mu)*rsig ; z_sig = sigmoid(o[:, 32:])
__global__ __launch_bounds__(256) void final_out(
    const float* __restrict__ o, const float* __restrict__ mu,
    const float* __restrict__ rs, float* __restrict__ out, int Nn) {
  int idx = blockIdx.x * 256 + threadIdx.x;
  if (idx >= Nn * 32) return;
  int r = idx >> 5, c = idx & 31;
  out[idx] = (o[(size_t)r * 64 + c] - mu[c]) * rs[c];
  float s = o[(size_t)r * 64 + 32 + c];
  out[(size_t)Nn * 32 + idx] = 1.f / (1.f + __expf(-s));
}

// ---------------- host ----------------

extern "C" void kernel_launch(void* const* d_in, const int* in_sizes, int n_in,
                              void* d_out, int out_size, void* d_ws, size_t ws_size,
                              hipStream_t stream) {
  const float* x    = (const float*)d_in[0];
  const int*   ei   = (const int*)d_in[1];     // edge_index [2,E] flat
  const float* W1l  = (const float*)d_in[2];
  const float* W1r  = (const float*)d_in[3];
  const float* att1 = (const float*)d_in[4];   // [4,64] flat
  // b1 (d_in[5]) cancels through BatchNorm -> unused
  const float* g1   = (const float*)d_in[6];
  const float* be1  = (const float*)d_in[7];
  const float* W2l  = (const float*)d_in[8];
  const float* W2r  = (const float*)d_in[9];
  const float* att2 = (const float*)d_in[10];
  const float* b2   = (const float*)d_in[11];
  float* out = (float*)d_out;

  const int Nn   = in_sizes[0] / F;            // 50000
  const int E    = in_sizes[1] / 2;            // 800000
  const int Etot = E + Nn;                     // +self loops

  // ---- carve workspace ----
  size_t off = 0;
  auto carve = [&](size_t bytes) -> void* {
    void* r = (void*)((char*)d_ws + off);
    off += (bytes + 255) & ~(size_t)255;
    return r;
  };
  __bf16* bt1l = (__bf16*)carve((size_t)F * F * 2);
  __bf16* bt1r = (__bf16*)carve((size_t)F * F * 2);
  __bf16* bt2l = (__bf16*)carve((size_t)F * F * 2);
  __bf16* bt2r = (__bf16*)carve((size_t)F * F * 2);
  float* XL    = (float*)carve((size_t)Nn * F * 4);
  float* XR    = (float*)carve((size_t)Nn * F * 4);
  float* AGG   = (float*)carve((size_t)Nn * F * 4);   // agg1 -> h -> agg2
  float* SCORE = (float*)carve((size_t)Etot * HH * 4);
  unsigned* NMAX = (unsigned*)carve((size_t)Nn * HH * 4);
  float* DEN   = (float*)carve((size_t)Nn * HH * 4);
  float* O     = (float*)carve((size_t)Nn * 64 * 4);
  float* SUM   = (float*)carve(256 * 4);
  float* SSQ   = (float*)carve(256 * 4);
  float* MU    = (float*)carve(256 * 4);
  float* RS    = (float*)carve(256 * 4);

  const int gW    = F * F / 256;               // 256 blocks
  const int gGemm = (Nn / 16 + 7) / 8;         // wave per 16-row strip
  const int gEdge = (Etot + 7) / 8;
  const int gEH   = (Etot * HH + 255) / 256;
  const int gNF   = (Nn * F + 255) / 256;
  const int gHM   = (Nn * 64 + 255) / 256;
  const int gFO   = (Nn * 32 + 255) / 256;

  // ---- weights -> bf16 transposed ----
  wconvert<<<gW, 256, 0, stream>>>(W1l, bt1l);
  wconvert<<<gW, 256, 0, stream>>>(W1r, bt1r);
  wconvert<<<gW, 256, 0, stream>>>(W2l, bt2l);
  wconvert<<<gW, 256, 0, stream>>>(W2r, bt2r);

  // ---- conv1 ----
  gemm_bf16<<<gGemm, 256, 0, stream>>>(x, bt1l, XL, Nn);
  gemm_bf16<<<gGemm, 256, 0, stream>>>(x, bt1r, XR, Nn);
  hipMemsetAsync(NMAX, 0, (size_t)Nn * HH * 4, stream);   // 0 < f2ord(-inf)
  hipMemsetAsync(DEN,  0, (size_t)Nn * HH * 4, stream);
  hipMemsetAsync(AGG,  0, (size_t)Nn * F * 4, stream);
  gat_score<<<gEdge, 256, 0, stream>>>(XL, XR, att1, ei, E, Etot, SCORE, NMAX);
  gat_expsum<<<gEH, 256, 0, stream>>>(ei, E, Etot, NMAX, SCORE, DEN);
  gat_aggregate<<<gEdge, 256, 0, stream>>>(ei, E, Etot, XL, SCORE, DEN, AGG);

  // ---- BN + ReLU (in place) ----
  hipMemsetAsync(SUM, 0, 256 * 4, stream);
  hipMemsetAsync(SSQ, 0, 256 * 4, stream);
  bn_partial<<<512, F, 0, stream>>>(AGG, Nn, F, SUM, SSQ);
  bn_finalize<<<1, F, 0, stream>>>(SUM, SSQ, Nn, F, MU, RS);
  bn_apply_relu<<<gNF, 256, 0, stream>>>(AGG, Nn * F, MU, RS, g1, be1);

  // ---- conv2 (same dims) ----
  gemm_bf16<<<gGemm, 256, 0, stream>>>(AGG, bt2l, XL, Nn);
  gemm_bf16<<<gGemm, 256, 0, stream>>>(AGG, bt2r, XR, Nn);
  hipMemsetAsync(NMAX, 0, (size_t)Nn * HH * 4, stream);
  hipMemsetAsync(DEN,  0, (size_t)Nn * HH * 4, stream);
  hipMemsetAsync(AGG,  0, (size_t)Nn * F * 4, stream);    // h no longer needed
  gat_score<<<gEdge, 256, 0, stream>>>(XL, XR, att2, ei, E, Etot, SCORE, NMAX);
  gat_expsum<<<gEH, 256, 0, stream>>>(ei, E, Etot, NMAX, SCORE, DEN);
  gat_aggregate<<<gEdge, 256, 0, stream>>>(ei, E, Etot, XL, SCORE, DEN, AGG);

  // ---- heads mean + b2, final BN / sigmoid ----
  head_mean<<<gHM, 256, 0, stream>>>(AGG, b2, O, Nn * 64);
  hipMemsetAsync(SUM, 0, 64 * 4, stream);
  hipMemsetAsync(SSQ, 0, 64 * 4, stream);
  bn_partial<<<512, 64, 0, stream>>>(O, Nn, 64, SUM, SSQ);
  bn_finalize<<<1, 64, 0, stream>>>(SUM, SSQ, Nn, 64, MU, RS);
  final_out<<<gFO, 256, 0, stream>>>(O, MU, RS, out, Nn);
}
